// BlockNonLocal_7318624272892
// MI455X (gfx1250) — matrix operand order repaired and checked by reading, not compiled, not run on hardware
//
#include <hip/hip_runtime.h>

#define Bb 2
#define Cc 64
#define Nn 8192   // D*H*W = 8*32*32

typedef __attribute__((ext_vector_type(16))) __bf16 v16bf;
typedef __attribute__((ext_vector_type(8)))  __bf16 v8bf;
typedef __attribute__((ext_vector_type(8)))  float  v8f;

__device__ __forceinline__ v16bf cat8(v8bf lo, v8bf hi) {
    v16bf r;
#pragma unroll
    for (int i = 0; i < 8; ++i) { r[i] = lo[i]; r[i + 8] = hi[i]; }
    return r;
}

__device__ __forceinline__ v8f wmma_bf16(v16bf a, v16bf b, v8f c) {
    return __builtin_amdgcn_wmma_f32_16x16x32_bf16(
        false, a, false, b, (short)0, c, false, false);
}

// ---- async global->LDS copy (CDNA5 VGLOBAL async path, ASYNCcnt-tracked) ----
__device__ __forceinline__ void async_ld_b128(unsigned lds_byte_off, unsigned gbyte_off,
                                              const void* sbase) {
    asm volatile("global_load_async_to_lds_b128 %0, %1, %2"
                 :: "v"(lds_byte_off), "v"(gbyte_off), "s"(sbase)
                 : "memory");
}

#if __has_builtin(__builtin_amdgcn_s_wait_asynccnt)
#define WAIT_ASYNC(n) __builtin_amdgcn_s_wait_asynccnt(n)
#else
#define WAIT_ASYNC(n) asm volatile("s_wait_asynccnt %0" :: "n"(n) : "memory")
#endif

// ---- lane-half (16-lane) row-max reduction on the VALU, not the DS pipe ----
#if __has_builtin(__builtin_amdgcn_mov_dpp8) && __has_builtin(__builtin_amdgcn_permlane16)
#define HAVE_FAST_SHUF 1
template<int M>
__device__ __forceinline__ float dpp8_xor(float v) {
    constexpr int sel = ((0 ^ M) & 7)        | (((1 ^ M) & 7) << 3)
                      | (((2 ^ M) & 7) << 6) | (((3 ^ M) & 7) << 9)
                      | (((4 ^ M) & 7) << 12)| (((5 ^ M) & 7) << 15)
                      | (((6 ^ M) & 7) << 18)| (((7 ^ M) & 7) << 21);
    return __builtin_bit_cast(float,
        __builtin_amdgcn_mov_dpp8(__builtin_bit_cast(int, v), sel));
}
__device__ __forceinline__ float perm_xor8(float v) {
    unsigned u = __builtin_bit_cast(unsigned, v);
    unsigned r = __builtin_amdgcn_permlane16(u, u, 0xFEDCBA98u, 0x76543210u, false, false);
    return __builtin_bit_cast(float, r);
}
#endif

__device__ __forceinline__ float rowmax16(float v) {
#ifdef HAVE_FAST_SHUF
    v = fmaxf(v, dpp8_xor<1>(v));
    v = fmaxf(v, dpp8_xor<2>(v));
    v = fmaxf(v, dpp8_xor<4>(v));
    v = fmaxf(v, perm_xor8(v));
#else
    v = fmaxf(v, __shfl_xor(v, 1, 32));
    v = fmaxf(v, __shfl_xor(v, 2, 32));
    v = fmaxf(v, __shfl_xor(v, 4, 32));
    v = fmaxf(v, __shfl_xor(v, 8, 32));
#endif
    return v;
}

// ---------------------------------------------------------------------------
// Kernel 0: one-time bf16 conversion of x into BOTH layouts so that every
// later tile staging is a contiguous-per-lane async b128 copy.
//   xc[b][c][n] = bf16(x[b][c][n])          (K/V view)
//   xt[b][n][c] = bf16(x[b][c][n])          (Q / K^T view)
// ---------------------------------------------------------------------------
__launch_bounds__(256)
__global__ void cvt_kernel(const float* __restrict__ x,
                           __bf16* __restrict__ xc, __bf16* __restrict__ xt)
{
    __shared__ __bf16 t[64 * 72];
    const int tid = threadIdx.x;
    const int b   = blockIdx.x / (Nn / 64);
    const int n0  = (blockIdx.x % (Nn / 64)) * 64;
    const float* xb  = x  + (size_t)b * Cc * Nn;
    __bf16*      xcb = xc + (size_t)b * Cc * Nn;
    __bf16*      xtb = xt + (size_t)b * Nn * Cc;

#pragma unroll
    for (int k = 0; k < 16; ++k) {
        int e = tid + k * 256;                // 64 c x 64 n
        int c = e >> 6, n = e & 63;
        __bf16 v = (__bf16)xb[(size_t)c * Nn + n0 + n];
        xcb[(size_t)c * Nn + n0 + n] = v;     // pass-through layout
        t[n * 72 + c] = v;                    // staged transpose
    }
    __syncthreads();
#pragma unroll
    for (int k = 0; k < 16; ++k) {
        int e = tid + k * 256;
        int n = e >> 6, c = e & 63;
        xtb[(size_t)(n0 + n) * Cc + c] = t[n * 72 + c];
    }
}

// ---------------------------------------------------------------------------
// Kernel 1: flash attention. Block = 256 thr = 8 waves, 128 query rows/block.
// All tile staging via global_load_async_to_lds_b128, K/V chunks double-
// buffered. Per 32-key chunk per wave: 4 WMMA (QK^T) + 4 WMMA (PV) +
// 1 WMMA (row-sums via ones matrix). Row-max reduction on VALU (dpp8 +
// permlane16) so the DS pipe only carries WMMA operands.
// ---------------------------------------------------------------------------
__launch_bounds__(256)
__global__ void nl_attn_kernel(const __bf16* __restrict__ xc,
                               const __bf16* __restrict__ xt,
                               float* __restrict__ y)
{
    __shared__ __bf16 q_lds[128 * 72];       // [n_local][c], stride 72 (144B rows)
    __shared__ __bf16 kt_lds[2 * 32 * 72];   // [buf][j][c]   (QK^T B-operands)
    __shared__ __bf16 kv_lds[2 * 64 * 32];   // [buf][c][j]   (PV  B-operands)
    __shared__ __bf16 p_lds[8 * 16 * 32];    // per-wave P re-layout buffer

    const int tid   = threadIdx.x;
    const int lane  = tid & 31;
    const int wave  = tid >> 5;
    const int lcol  = lane & 15;
    const int lhalf = lane >> 4;
    const int b     = blockIdx.x / (Nn / 128);
    const int n0    = (blockIdx.x % (Nn / 128)) * 128;
    const __bf16* xcb = xc + (size_t)b * Cc * Nn;
    const __bf16* xtb = xt + (size_t)b * Nn * Cc;

    const unsigned q_base  = (unsigned)(unsigned long long)(void*)q_lds;
    const unsigned kt_base = (unsigned)(unsigned long long)(void*)kt_lds;
    const unsigned kv_base = (unsigned)(unsigned long long)(void*)kv_lds;

    // --- issue Q block (128 rows x 128B) : 4 async b128 per thread ---
#pragma unroll
    for (int k = 0; k < 4; ++k) {
        int u = tid + k * 256;                // 1024 x 16B units
        int n = u >> 3, w = u & 7;
        async_ld_b128(q_base + n * 144 + w * 16,
                      (unsigned)((size_t)(n0 + n) * 128) + w * 16, xtb);
    }
    // --- issue chunk 0 into buffer 0 (1 kv unit + 1 kt unit per thread) ---
    {
        int c = tid >> 2, part = tid & 3;     // kv: 64 rows x 64B
        async_ld_b128(kv_base + c * 64 + part * 16,
                      (unsigned)((size_t)c * Nn * 2) + part * 16, xcb);
        int j = tid >> 3, w = tid & 7;        // kt: 32 rows x 128B
        async_ld_b128(kt_base + j * 144 + w * 16,
                      (unsigned)((size_t)j * 128) + w * 16, xtb);
    }
    WAIT_ASYNC(2);            // in-order: Q done (chunk0 may still be inflight)
    __syncthreads();

    // Q A-operands held in VGPRs for the whole key loop.
    const int qrow = wave * 16 + lcol;
    const int hq   = lhalf * 8;
    v16bf aQ0 = cat8(*(const v8bf*)&q_lds[qrow * 72 + lhalf * 16],
                     *(const v8bf*)&q_lds[qrow * 72 + lhalf * 16 + 8]);
    v16bf aQ1 = cat8(*(const v8bf*)&q_lds[qrow * 72 + 32 + lhalf * 16],
                     *(const v8bf*)&q_lds[qrow * 72 + 32 + lhalf * 16 + 8]);

    v16bf bOnes;
#pragma unroll
    for (int i = 0; i < 16; ++i) bOnes[i] = (__bf16)1.0f;

    v8f o0 = {}, o1 = {}, o2 = {}, o3 = {}, lt = {};
    float mrow[8];
#pragma unroll
    for (int i = 0; i < 8; ++i) mrow[i] = -1e30f;

    int pp = 0;
    for (int j0 = 0; j0 < Nn; j0 += 32) {
        const int nj = j0 + 32;
        if (nj < Nn) {        // prefetch next chunk into the other buffer
            const int np = pp ^ 1;
            int c = tid >> 2, part = tid & 3;
            async_ld_b128(kv_base + np * 4096 + c * 64 + part * 16,
                          (unsigned)(((size_t)c * Nn + nj) * 2) + part * 16, xcb);
            int j = tid >> 3, w = tid & 7;
            async_ld_b128(kt_base + np * 4608 + j * 144 + w * 16,
                          (unsigned)((size_t)(nj + j) * 128) + w * 16, xtb);
            WAIT_ASYNC(2);    // current chunk's (older) copies complete
        } else {
            WAIT_ASYNC(0);
        }
        __syncthreads();

        const __bf16* kt = kt_lds + pp * (32 * 72);
        const __bf16* kv = kv_lds + pp * (64 * 32);

        // S[16x32] = Q[16x64] * K[64x32] : two key tiles, K-dim split 2x32.
        v8f s0 = {}, s1 = {};
        {
            int jr = lcol;
            v16bf bk0 = cat8(*(const v8bf*)&kt[jr * 72 + 0  + lhalf * 16],
                             *(const v8bf*)&kt[jr * 72 + 8  + lhalf * 16]);
            v16bf bk1 = cat8(*(const v8bf*)&kt[jr * 72 + 32 + lhalf * 16],
                             *(const v8bf*)&kt[jr * 72 + 40 + lhalf * 16]);
            s0 = wmma_bf16(aQ0, bk0, s0);
            s0 = wmma_bf16(aQ1, bk1, s0);
        }
        {
            int jr = 16 + lcol;
            v16bf bk0 = cat8(*(const v8bf*)&kt[jr * 72 + 0  + lhalf * 16],
                             *(const v8bf*)&kt[jr * 72 + 8  + lhalf * 16]);
            v16bf bk1 = cat8(*(const v8bf*)&kt[jr * 72 + 32 + lhalf * 16],
                             *(const v8bf*)&kt[jr * 72 + 40 + lhalf * 16]);
            s1 = wmma_bf16(aQ0, bk0, s1);
            s1 = wmma_bf16(aQ1, bk1, s1);
        }

        // Online softmax. comp i <-> row (lhalf*8 + i) in C-layout.
        float p0[8], p1[8];
#pragma unroll
        for (int i = 0; i < 8; ++i) {
            float v  = rowmax16(fmaxf(s0[i], s1[i]));
            float mn = fmaxf(mrow[i], v);
            float sc = __expf(mrow[i] - mn);
            mrow[i] = mn;
            p0[i] = __expf(s0[i] - mn);
            p1[i] = __expf(s1[i] - mn);
            o0[i] *= sc; o1[i] *= sc; o2[i] *= sc; o3[i] *= sc; lt[i] *= sc;
        }

        // P (C-format) -> A-format via per-wave LDS scratch.
        const int pb = wave * 512;
#pragma unroll
        for (int i = 0; i < 8; ++i) {
            int row = lhalf * 8 + i;
            p_lds[pb + row * 32 + lcol]      = (__bf16)p0[i];
            p_lds[pb + row * 32 + 16 + lcol] = (__bf16)p1[i];
        }
        v16bf aP = cat8(*(const v8bf*)&p_lds[pb + lcol * 32 + lhalf * 16],
                        *(const v8bf*)&p_lds[pb + lcol * 32 + lhalf * 16 + 8]);

        // O[16x64] += P * V (4 channel tiles); row-sums via ones-matrix WMMA.
        v16bf bv0 = cat8(*(const v8bf*)&kv[(0  + lcol) * 32 + lhalf * 16],
                         *(const v8bf*)&kv[(0  + lcol) * 32 + lhalf * 16 + 8]);
        v16bf bv1 = cat8(*(const v8bf*)&kv[(16 + lcol) * 32 + lhalf * 16],
                         *(const v8bf*)&kv[(16 + lcol) * 32 + lhalf * 16 + 8]);
        v16bf bv2 = cat8(*(const v8bf*)&kv[(32 + lcol) * 32 + lhalf * 16],
                         *(const v8bf*)&kv[(32 + lcol) * 32 + lhalf * 16 + 8]);
        v16bf bv3 = cat8(*(const v8bf*)&kv[(48 + lcol) * 32 + lhalf * 16],
                         *(const v8bf*)&kv[(48 + lcol) * 32 + lhalf * 16 + 8]);
        o0 = wmma_bf16(aP, bv0, o0);
        o1 = wmma_bf16(aP, bv1, o1);
        o2 = wmma_bf16(aP, bv2, o2);
        o3 = wmma_bf16(aP, bv3, o3);
        lt = wmma_bf16(aP, bOnes, lt);   // l_row += sum_j P[row][j]

        __syncthreads();
        pp ^= 1;
    }

    // Normalize and write y[b][n][c] (f32).
    float* yb = y + (size_t)b * Nn * Cc;
#pragma unroll
    for (int i = 0; i < 8; ++i) {
        int n = n0 + wave * 16 + lhalf * 8 + i;
        float inv = 1.0f / lt[i];
        yb[(size_t)n * Cc + 0  + lcol] = o0[i] * inv;
        yb[(size_t)n * Cc + 16 + lcol] = o1[i] * inv;
        yb[(size_t)n * Cc + 32 + lcol] = o2[i] * inv;
        yb[(size_t)n * Cc + 48 + lcol] = o3[i] * inv;
    }
}

// ---------------------------------------------------------------------------
// Kernel 2: z[b,o,n] = sum_c w[o,c]*y[b,n,c] + bias[o] + x[b,o,n]
// Memory-bound (12 MB @ 23.3 TB/s ~ 0.5us); f32 VALU, coalesced over n.
// ---------------------------------------------------------------------------
__launch_bounds__(256)
__global__ void wmap_res_kernel(const float* __restrict__ y, const float* __restrict__ w,
                                const float* __restrict__ bias, const float* __restrict__ x,
                                float* __restrict__ z)
{
    __shared__ float wsh[64 * 64];
    __shared__ float ysh[64 * 65];

    const int tid = threadIdx.x;
    const int b   = blockIdx.x / (Nn / 64);
    const int n0  = (blockIdx.x % (Nn / 64)) * 64;

#pragma unroll
    for (int k = 0; k < 16; ++k) { int e = tid + k * 256; wsh[e] = w[e]; }
    const float* yb = y + ((size_t)b * Nn + n0) * Cc;
#pragma unroll
    for (int k = 0; k < 16; ++k) {
        int e = tid + k * 256;
        int n = e >> 6, c = e & 63;
        ysh[n * 65 + c] = yb[(size_t)n * Cc + c];
    }
    __syncthreads();

    const int n  = tid & 63;
    const int ob = tid >> 6;
    float acc[16];
#pragma unroll
    for (int k = 0; k < 16; ++k) acc[k] = 0.0f;

    for (int c = 0; c < 64; ++c) {
        float yv = ysh[n * 65 + c];
#pragma unroll
        for (int k = 0; k < 16; ++k)
            acc[k] += wsh[(ob * 16 + k) * 64 + c] * yv;
    }

    const float* xb = x + (size_t)b * Cc * Nn;
    float*       zb = z + (size_t)b * Cc * Nn;
#pragma unroll
    for (int k = 0; k < 16; ++k) {
        int o = ob * 16 + k;
        size_t idx = (size_t)o * Nn + n0 + n;
        zb[idx] = acc[k] + bias[o] + xb[idx];
    }
}

extern "C" void kernel_launch(void* const* d_in, const int* in_sizes, int n_in,
                              void* d_out, int out_size, void* d_ws, size_t ws_size,
                              hipStream_t stream)
{
    const float* x    = (const float*)d_in[0];   // [B, C, D, H, W]
    const float* w    = (const float*)d_in[1];   // [C, C]
    const float* bias = (const float*)d_in[2];   // [C]
    float* z = (float*)d_out;

    char* ws = (char*)d_ws;
    __bf16* xc = (__bf16*)ws;                                   // 2 MB
    __bf16* xt = (__bf16*)(ws + (size_t)Bb * Cc * Nn * 2);      // 2 MB
    float*  y  = (float*) (ws + (size_t)Bb * Cc * Nn * 4);      // 4 MB

    cvt_kernel   <<<Bb * (Nn / 64),  256, 0, stream>>>(x, xc, xt);
    nl_attn_kernel<<<Bb * (Nn / 128), 256, 0, stream>>>(xc, xt, y);
    wmap_res_kernel<<<Bb * (Nn / 64), 256, 0, stream>>>(y, w, bias, x, z);
}
